// TritonSegNCA_4861902979111
// MI455X (gfx1250) — matrix operand order repaired and checked by reading, not compiled, not run on hardware
//
#include <hip/hip_runtime.h>
#include <hip/hip_bf16.h>

// ---------------------------------------------------------------------------
// NCA on MI455X (gfx1250): 32 fused step kernels. State (8,256,256,16) f32 =
// 33.5MB lives in L2 (192MB). Per-pixel MLP 48->128->16 done with
// v_wmma_f32_16x16x32_f16 (f16 in, f32 accumulate). Depthwise 3x3 convs and
// mask in VALU. Halo tile staged with GLOBAL_LOAD_ASYNC_TO_LDS_B128
// (ASYNCcnt). Ping-pong: bufA in d_ws, bufB aliases d_out (finalize reads
// only bufA -> no overlap hazard).
// ---------------------------------------------------------------------------

typedef __attribute__((ext_vector_type(16))) _Float16 f16x16;
typedef __attribute__((ext_vector_type(8)))  _Float16 f16x8;
typedef __attribute__((ext_vector_type(8)))  float    f32x8;

#define NCA_B     8
#define NCA_HW    256
#define NCA_C     16
#define NCA_HID   128
#define NCA_STEPS 32
#define NHWC_ELEMS (NCA_B * NCA_HW * NCA_HW * NCA_C)   // 8388608
#define BHW_ELEMS  (NCA_B * NCA_HW * NCA_HW)           // 524288

static __device__ __forceinline__ f32x8 wmma_f16(f16x16 a, f16x16 b, f32x8 c) {
  // D = A(16x32 f16) * B(32x16 f16) + C(16x16 f32)
  return __builtin_amdgcn_wmma_f32_16x16x32_f16(
      /*neg_a=*/false, a, /*neg_b=*/false, b,
      /*c_mod=*/(short)0, c, /*reuse_a=*/false, /*reuse_b=*/false);
}

// A-fragment (16-bit A 16x32), contiguous K order to match B: per lane,
// halves 0..15 are K = kbase..kbase+15.
static __device__ __forceinline__ f16x16 load_afrag(const _Float16* p) {
  f16x8 lo = *(const f16x8*)(p);
  f16x8 hi = *(const f16x8*)(p + 8);
  f16x16 a;
#pragma unroll
  for (int i = 0; i < 8; ++i) { a[i] = lo[i]; a[i + 8] = hi[i]; }
  return a;
}

// Flat->LDS aperture mapping (ISA 10.2): the
// low 32 bits of a generic pointer to __shared__ data are the LDS address.
static __device__ __forceinline__ unsigned lds_addr32(const void* p) {
  return (unsigned)(unsigned long long)p;
}

// Async copy global -> LDS, 16 bytes per active lane. Tracked by ASYNCcnt.
static __device__ __forceinline__ void async_copy_b128(unsigned lds_dst,
                                                       const float* gsrc) {
  asm volatile("global_load_async_to_lds_b128 %0, %1, off"
               :
               : "v"(lds_dst), "v"(gsrc)
               : "memory");
}
static __device__ __forceinline__ void wait_asynccnt0() {
  asm volatile("s_wait_asynccnt 0x0" ::: "memory");
}

// Threefry-2x32-20 keyed on (42, step), counter (pixel, const). Deterministic;
// not bit-identical to JAX's key-split stream (no runtime validation exists).
static __device__ __forceinline__ float nca_uniform(unsigned step, unsigned ctr) {
  unsigned ks[3];
  ks[0] = 42u;
  ks[1] = step;
  ks[2] = 0x1BD11BDAu ^ ks[0] ^ ks[1];
  unsigned x0 = ctr + ks[0];
  unsigned x1 = 0x3243F6A8u + ks[1];
  const unsigned rotA[4] = {13u, 15u, 26u, 6u};
  const unsigned rotB[4] = {17u, 29u, 16u, 24u};
#pragma unroll
  for (int i = 0; i < 5; ++i) {
    const unsigned* rot = (i & 1) ? rotB : rotA;
#pragma unroll
    for (int j = 0; j < 4; ++j) {
      x0 += x1;
      x1 = (x1 << rot[j]) | (x1 >> (32u - rot[j]));
      x1 ^= x0;
    }
    x0 += ks[(i + 1) % 3];
    x1 += ks[(i + 2) % 3] + (unsigned)(i + 1);
  }
  return (float)(x0 >> 8) * (1.0f / 16777216.0f);
}

// --------------------------- layout kernels --------------------------------

__global__ void nchw_to_nhwc(const float* __restrict__ in, float* __restrict__ out) {
  int i = blockIdx.x * blockDim.x + threadIdx.x;           // NHWC flat id
  int c = i & 15;
  int x = (i >> 4) & 255;
  int y = (i >> 12) & 255;
  int b = i >> 20;
  out[i] = in[(((b * NCA_C + c) * NCA_HW + y) * NCA_HW) + x];
}

// Pre-transpose MLP weights to f16 [N][K] so each WMMA B-fragment lane reads
// 16 contiguous halves (one 32B load -> 2x global_load_b128).
__global__ void prep_weights(const float* __restrict__ w_fc0,
                             const float* __restrict__ w_fc1,
                             _Float16* __restrict__ wb0,   // [128][64], K padded 48->64
                             _Float16* __restrict__ wb1) { // [16][128]
  int tid = threadIdx.x;
  for (int i = tid; i < NCA_HID * 64; i += 128) {
    int n = i >> 6, k = i & 63;
    wb0[i] = (k < 48) ? (_Float16)w_fc0[k * NCA_HID + n] : (_Float16)0.f;
  }
  for (int i = tid; i < NCA_C * NCA_HID; i += 128) {
    int n = i >> 7, k = i & 127;
    wb1[i] = (_Float16)w_fc1[k * NCA_C + n];
  }
}

// ------------------------------ step kernel --------------------------------
// Block: 128 threads (4 waves), tile = 16 wide x 8 tall pixels.
// LDS: halo tile (f32), perception (f16, K padded to 64), hidden h (f16),
// conv weights, bias, mask. Total 62,848 B.

__global__ __launch_bounds__(128) void nca_step(
    const float* __restrict__ src, float* __restrict__ dst,
    const float* __restrict__ wp0, const float* __restrict__ wp1,
    const float* __restrict__ bias,
    const _Float16* __restrict__ wb0, const _Float16* __restrict__ wb1,
    int step) {
  __shared__ __align__(16) float    sxt[10 * 18 * 16];          // 11520 B
  __shared__ __align__(32) _Float16 sperc[128 * 64];            // 16384 B
  __shared__ __align__(32) _Float16 sHid[128 * 128];            // 32768 B
  __shared__ float sw0[144], sw1[144], sbias[128], smask[128];  //  2176 B

  const int tid  = threadIdx.x;
  const int lane = tid & 31;
  const int wave = tid >> 5;
  const int bimg = blockIdx.z;
  const int gy0  = blockIdx.y * 8;
  const int gx0  = blockIdx.x * 16;

  sbias[tid] = bias[tid];
  for (int i = tid; i < 144; i += 128) { sw0[i] = wp0[i]; sw1[i] = wp1[i]; }

  // Halo tile (18x10 pixels x 16ch, reflect pad) via async global->LDS b128:
  // one pixel-quad (4 floats) per active lane, no VGPR data round-trip.
  for (int j = tid; j < 10 * 18 * 4; j += 128) {   // 720 16B chunks
    int quad = j & 3;
    int pix  = j >> 2;
    int xx   = pix % 18;
    int yy   = pix / 18;
    int gy = gy0 + yy - 1; gy = (gy < 0) ? 1 : (gy > 255 ? 254 : gy);
    int gx = gx0 + xx - 1; gx = (gx < 0) ? 1 : (gx > 255 ? 254 : gx);
    const float* g = src + (((bimg * NCA_HW + gy) * NCA_HW + gx) << 4) + quad * 4;
    async_copy_b128(lds_addr32(&sxt[pix * 16 + quad * 4]), g);
  }
  wait_asynccnt0();
  __syncthreads();

  // Perception (one thread per pixel): [x | dwconv0 | dwconv1 | 0-pad] -> f16.
  {
    const int px = tid & 15, py = tid >> 4;   // py in 0..7
    const int p = tid;
#pragma unroll
    for (int c = 0; c < NCA_C; ++c) {
      float s0 = 0.f, s1 = 0.f;
#pragma unroll
      for (int ky = 0; ky < 3; ++ky)
#pragma unroll
        for (int kx = 0; kx < 3; ++kx) {
          float v = sxt[(((py + ky) * 18) + (px + kx)) * 16 + c];
          s0 = fmaf(v, sw0[(ky * 3 + kx) * 16 + c], s0);
          s1 = fmaf(v, sw1[(ky * 3 + kx) * 16 + c], s1);
        }
      sperc[p * 64 + c]      = (_Float16)sxt[(((py + 1) * 18) + (px + 1)) * 16 + c];
      sperc[p * 64 + 16 + c] = (_Float16)s0;
      sperc[p * 64 + 32 + c] = (_Float16)s1;
      sperc[p * 64 + 48 + c] = (_Float16)0.f;
    }
    unsigned pix = ((unsigned)bimg << 16) | ((unsigned)(gy0 + py) << 8) | (unsigned)(gx0 + px);
    smask[p] = (nca_uniform((unsigned)step, pix) <= 0.5f) ? 1.f : 0.f;
  }
  __syncthreads();

  const int r      = lane & 15;     // A row / B-C column within tile
  const int hiHalf = lane >> 4;     // 0: lanes 0-15, 1: lanes 16-31
  const int kb     = hiHalf * 16;   // A-fragment K sub-base per lane half
  const int t0     = wave * 2;      // this wave's two M-tiles (tile rows)

  // GEMM1: h = relu(perc(16x64) @ w_fc0(64x128) + b).
  // A fragments hoisted for both M-tiles; B fragments loaded once per N-tile
  // and reused across both M-tiles.
  {
    const _Float16* arow0 = &sperc[((t0 + 0) * 16 + r) * 64];
    const _Float16* arow1 = &sperc[((t0 + 1) * 16 + r) * 64];
    f16x16 a00 = load_afrag(arow0 + 0  + kb);   // tile 0, K 0..31
    f16x16 a01 = load_afrag(arow0 + 32 + kb);   // tile 0, K 32..63
    f16x16 a10 = load_afrag(arow1 + 0  + kb);   // tile 1, K 0..31
    f16x16 a11 = load_afrag(arow1 + 32 + kb);   // tile 1, K 32..63
#pragma unroll
    for (int n = 0; n < 8; ++n) {
      const _Float16* bp = wb0 + (n * 16 + r) * 64;   // B lane: N = n*16+r
      f16x16 b0 = *(const f16x16*)(bp + hiHalf * 16);        // K 0..31
      f16x16 b1 = *(const f16x16*)(bp + 32 + hiHalf * 16);   // K 32..63
      f32x8 c0 = {}, c1 = {};
      c0 = wmma_f16(a00, b0, c0);
      c0 = wmma_f16(a01, b1, c0);
      c1 = wmma_f16(a10, b0, c1);
      c1 = wmma_f16(a11, b1, c1);
      float bv = sbias[n * 16 + r];
#pragma unroll
      for (int q = 0; q < 8; ++q) {                   // C: M = q + 8*hiHalf
        int px = q + hiHalf * 8;
        float h0 = fmaxf(c0[q] + bv, 0.f);
        float h1 = fmaxf(c1[q] + bv, 0.f);
        sHid[((t0 + 0) * 16 + px) * 128 + n * 16 + r] = (_Float16)h0;
        sHid[((t0 + 1) * 16 + px) * 128 + n * 16 + r] = (_Float16)h1;
      }
    }
  }
  __syncthreads();

  // GEMM2: dx = h(16x128) @ w_fc1(128x16); B fragment shared by both M-tiles.
  {
    f32x8 acc0 = {}, acc1 = {};
#pragma unroll
    for (int kc = 0; kc < 4; ++kc) {
      f16x16 b = *(const f16x16*)(wb1 + r * 128 + kc * 32 + hiHalf * 16);
      f16x16 a0 = load_afrag(&sHid[((t0 + 0) * 16 + r) * 128 + kc * 32 + kb]);
      f16x16 a1 = load_afrag(&sHid[((t0 + 1) * 16 + r) * 128 + kc * 32 + kb]);
      acc0 = wmma_f16(a0, b, acc0);
      acc1 = wmma_f16(a1, b, acc1);
    }
    // Residual + stochastic mask + channel pinning; lanes 0-15 / 16-31 each
    // cover 16 contiguous channels of one pixel -> coalesced 64B segments.
#pragma unroll
    for (int mi = 0; mi < 2; ++mi) {
      const int t = t0 + mi;
      const int gy = gy0 + t;
      const f32x8& acc = mi ? acc1 : acc0;
#pragma unroll
      for (int q = 0; q < 8; ++q) {
        int px = q + hiHalf * 8;                      // pixel column in tile
        int ch = r;                                   // channel
        float xc = sxt[(((t + 1) * 18) + (px + 1)) * 16 + ch];
        float m  = smask[t * 16 + px];
        float v  = (ch < 3) ? xc : fmaf(acc[q], m, xc);
        dst[(((bimg * NCA_HW + gy) * NCA_HW + (gx0 + px)) << 4) + ch] = v;
      }
    }
  }
}

// ------------------------------ finalize -----------------------------------
// d_out = [ x_final[:,3,:,:] (BHW) | x_final NCHW (BCHW) ], reading only bufA.

__global__ void finalize_out(const float* __restrict__ nhwc, float* __restrict__ out) {
  int i = blockIdx.x * blockDim.x + threadIdx.x;     // NCHW flat id
  int x = i & 255;
  int y = (i >> 8) & 255;
  int c = (i >> 16) & 15;
  int b = i >> 20;
  float v = nhwc[(((b * NCA_HW + y) * NCA_HW + x) << 4) + c];
  out[BHW_ELEMS + i] = v;
  if (c == 3) out[(b << 16) | (y << 8) | x] = v;
}

// ------------------------------- host --------------------------------------

extern "C" void kernel_launch(void* const* d_in, const int* in_sizes, int n_in,
                              void* d_out, int out_size, void* d_ws, size_t ws_size,
                              hipStream_t stream) {
  (void)in_sizes; (void)n_in; (void)out_size; (void)ws_size;
  const float* x     = (const float*)d_in[0];
  const float* w_p0  = (const float*)d_in[1];
  const float* w_p1  = (const float*)d_in[2];
  const float* w_fc0 = (const float*)d_in[3];
  const float* b_fc0 = (const float*)d_in[4];
  const float* w_fc1 = (const float*)d_in[5];

  float*    bufA = (float*)d_ws;                                   // 33.55 MB
  _Float16* wb0  = (_Float16*)((char*)d_ws + (size_t)NHWC_ELEMS * 4);
  _Float16* wb1  = wb0 + NCA_HID * 64;
  float*    bufB = (float*)d_out;   // 35.6 MB >= 33.55 MB; finalize reads bufA only

  nchw_to_nhwc<<<NHWC_ELEMS / 256, 256, 0, stream>>>(x, bufA);
  prep_weights<<<1, 128, 0, stream>>>(w_fc0, w_fc1, wb0, wb1);

  dim3 grid(NCA_HW / 16, NCA_HW / 8, NCA_B);   // 16 x 32 x 8 blocks
  for (int s = 0; s < NCA_STEPS; ++s) {
    const float* src = (s & 1) ? bufB : bufA;
    float*       dst = (s & 1) ? bufA : bufB;
    nca_step<<<grid, 128, 0, stream>>>(src, dst, w_p0, w_p1, b_fc0, wb0, wb1, s);
  }
  // 32 steps (even count) -> final state in bufA.
  finalize_out<<<NHWC_ELEMS / 256, 256, 0, stream>>>(bufA, (float*)d_out);
}
